// HyperGRUCell_14886356648164
// MI455X (gfx1250) — matrix operand order repaired and checked by reading, not compiled, not run on hardware
//
#include <hip/hip_runtime.h>

typedef __attribute__((ext_vector_type(16))) __bf16 v16bf;
typedef __attribute__((ext_vector_type(8)))  __bf16 v8bf;
typedef __attribute__((ext_vector_type(8)))  float  v8f;
typedef unsigned short ushort_t;

#define EPSF   1e-15f
#define CLIPF  0.99999f   // 1 - 1e-5

// ---------------------------------------------------------------- helpers

__device__ inline ushort_t f2bf_rne(float f) {
    unsigned u = __float_as_uint(f);
    unsigned r = u + 0x7FFFu + ((u >> 16) & 1u);
    return (ushort_t)(r >> 16);
}

__device__ inline void load4(float* d, const float* p) {
    float4 v = *(const float4*)p;
    d[0] = v.x; d[1] = v.y; d[2] = v.z; d[3] = v.w;
}
__device__ inline void store4(float* p, const float* d) {
    float4 v; v.x = d[0]; v.y = d[1]; v.z = d[2]; v.w = d[3];
    *(float4*)p = v;
}

__device__ inline float dot4(const float* a, const float* b) {
    float s = 0.f;
#pragma unroll
    for (int i = 0; i < 4; ++i) s += a[i] * b[i];
    return s;
}

// block = 256 threads = 8 wave32
__device__ inline float blockSum(float v, float* smem) {
    const int lane = threadIdx.x & 31;
    const int w    = threadIdx.x >> 5;
#pragma unroll
    for (int o = 16; o > 0; o >>= 1) v += __shfl_down(v, o, 32);
    if (lane == 0) smem[w] = v;
    __syncthreads();
    if (w == 0) {
        float t = (lane < 8) ? smem[lane] : 0.f;
#pragma unroll
        for (int o = 4; o > 0; o >>= 1) t += __shfl_down(t, o, 32);
        if (lane == 0) smem[0] = t;
    }
    __syncthreads();
    float s = smem[0];
    __syncthreads();
    return s;
}

__device__ inline float atanh_clip(float z) {
    z = fminf(fmaxf(z, -CLIPF), CLIPF);
    return atanhf(z);
}

// res = tanh(|mx|/xn * atanh(xn)) * mx/|mx|   (0 if |mx| ~ 0)
__device__ inline void matvec_scale4(const float* mx, float xn, float* o, float* smem) {
    float raw = sqrtf(blockSum(dot4(mx, mx), smem));
    float vn  = fmaxf(raw, EPSF);
    float t   = tanhf(vn / xn * atanh_clip(xn));
    float sc  = (raw <= EPSF) ? 0.f : (t / vn);
#pragma unroll
    for (int i = 0; i < 4; ++i) o[i] = mx[i] * sc;
}

__device__ inline void mobius_add4(const float* a, const float* b, float* o, float* smem) {
    float x2 = blockSum(dot4(a, a), smem);
    float y2 = blockSum(dot4(b, b), smem);
    float xy = blockSum(dot4(a, b), smem);
    float ca  = 1.f + 2.f * xy + y2;
    float cb  = 1.f - x2;
    float den = fmaxf(1.f + 2.f * xy + x2 * y2, EPSF);
    float inv = 1.f / den;
#pragma unroll
    for (int i = 0; i < 4; ++i) o[i] = (ca * a[i] + cb * b[i]) * inv;
}

__device__ inline void logmap0_4(const float* y, float* o, float* smem) {
    float raw = sqrtf(blockSum(dot4(y, y), smem));
    float yn  = fmaxf(raw, EPSF);
    float sc  = atanh_clip(yn) / yn;
#pragma unroll
    for (int i = 0; i < 4; ++i) o[i] = y[i] * sc;
}

// mobius_pointwise_mul(w, x); returns ||result|| (0 if degenerate)
__device__ inline float pointwise_mul4(const float* w, const float* x, float* o, float* smem) {
    float xn = fmaxf(sqrtf(blockSum(dot4(x, x), smem)), EPSF);
    float wx[4];
#pragma unroll
    for (int i = 0; i < 4; ++i) wx[i] = w[i] * x[i];
    float raw = sqrtf(blockSum(dot4(wx, wx), smem));
    float wxn = fmaxf(raw, EPSF);
    float t   = tanhf(wxn / xn * atanh_clip(xn));
    float sc  = (raw <= EPSF) ? 0.f : (t / wxn);
#pragma unroll
    for (int i = 0; i < 4; ++i) o[i] = wx[i] * sc;
    return (raw <= EPSF) ? 0.f : t;
}

// ---------------------------------------------------------------- convert

__global__ __launch_bounds__(256) void cvt_f32_bf16(const float* __restrict__ in,
                                                    ushort_t* __restrict__ out, int n) {
    int i = blockIdx.x * 256 + threadIdx.x;
    if (i < n) out[i] = f2bf_rne(in[i]);
}

// ---------------------------------------------------------------- WMMA GEMM
// C[m,n] = sum_k A[m,k] * W[n,k]   (A: Mx1024 row-major bf16, W: 1024x1024 row-major bf16)
// block = 256 thr = 8 waves; wave tile 64x64 (4x4 WMMA); block tile 256(M) x 128(N)
// N and K are compile-time so sub-tile strides fold into 24-bit instruction offsets.

#define GK 1024
#define GN 1024

__device__ inline v16bf load_a_frag(const ushort_t* p) {
    v8bf lo = *(const v8bf*)(p);
    v8bf hi = *(const v8bf*)(p + 16);
    return __builtin_shufflevector(lo, hi, 0, 1, 2, 3, 4, 5, 6, 7,
                                           8, 9, 10, 11, 12, 13, 14, 15);
}

__global__ __launch_bounds__(256) void gemm_bf16_nt(const ushort_t* __restrict__ A,
                                                    const ushort_t* __restrict__ W,
                                                    float* __restrict__ C) {
    const int lane = threadIdx.x & 31;
    const int wave = threadIdx.x >> 5;
    const int wm   = wave & 3;       // 0..3  -> M
    const int wn   = wave >> 2;      // 0..1  -> N
    const int mW   = blockIdx.y * 256 + wm * 64;
    const int nW   = blockIdx.x * 128 + wn * 64;
    const int r    = lane & 15;
    const int half = lane >> 4;

    v8f acc[4][4];
#pragma unroll
    for (int i = 0; i < 4; ++i)
#pragma unroll
        for (int j = 0; j < 4; ++j)
            acc[i][j] = (v8f){0.f,0.f,0.f,0.f,0.f,0.f,0.f,0.f};

    const ushort_t* aP = A + (size_t)(mW + r) * GK;   // sub-tile i at +i*16*GK (const offset)
    const ushort_t* bP = W + (size_t)(nW + r) * GK;   // sub-tile j at +j*16*GK (const offset)

    for (int k0 = 0; k0 < GK; k0 += 32) {
        const int ka = k0 + half * 8;    // A frag: K 0-7 & 16-23 (lane<16), 8-15 & 24-31 (lane>=16)
        const int kb = k0 + half * 16;   // unused: B frag uses the same K order as A (ka)
        v16bf a[4], b[4];
#pragma unroll
        for (int i = 0; i < 4; ++i) a[i] = load_a_frag(aP + ka + i * 16 * GK);
#pragma unroll
        for (int j = 0; j < 4; ++j) b[j] = load_a_frag(bP + ka + j * 16 * GK);
#pragma unroll
        for (int i = 0; i < 4; ++i)
#pragma unroll
            for (int j = 0; j < 4; ++j)
                acc[i][j] = __builtin_amdgcn_wmma_f32_16x16x32_bf16(
                    false, a[i], false, b[j], (short)0, acc[i][j], false, false);
    }

    // C/D layout: lanes 0-15 hold N=lane, VGPR e -> M=e; lanes 16-31 hold N=lane-16, VGPR e -> M=e+8
    float* cBase = C + (size_t)(mW + half * 8) * GN + nW + r;
#pragma unroll
    for (int i = 0; i < 4; ++i)
#pragma unroll
        for (int j = 0; j < 4; ++j)
#pragma unroll
            for (int e = 0; e < 8; ++e)
                cBase[(size_t)(i * 16 + e) * GN + j * 16] = acc[i][j][e];
}

// ---------------------------------------------------------------- gates (z, r, r*h)

__global__ __launch_bounds__(256) void gates_kernel(
        const float* __restrict__ x, const float* __restrict__ hid,
        const float* __restrict__ whz, const float* __restrict__ uxz,
        const float* __restrict__ whr, const float* __restrict__ uxr,
        const float* __restrict__ bz, const float* __restrict__ br,
        float* __restrict__ z_out, ushort_t* __restrict__ rph_bf,
        float* __restrict__ xnorm, float* __restrict__ hidnorm,
        float* __restrict__ rphnorm) {
    __shared__ float smem[8];
    const int    b    = blockIdx.x;
    const size_t base = (size_t)b * 1024;
    const int    i0   = threadIdx.x * 4;

    float xv[4], hv[4], wzv[4], uzv[4], wrv[4], urv[4], bzv[4], brv[4];
    load4(xv,  x   + base + i0);
    load4(hv,  hid + base + i0);
    load4(wzv, whz + base + i0);
    load4(uzv, uxz + base + i0);
    load4(wrv, whr + base + i0);
    load4(urv, uxr + base + i0);
    load4(bzv, bz + i0);
    load4(brv, br + i0);

    float xn = fmaxf(sqrtf(blockSum(dot4(xv, xv), smem)), EPSF);
    float hn = fmaxf(sqrtf(blockSum(dot4(hv, hv), smem)), EPSF);
    if (threadIdx.x == 0) { xnorm[b] = xn; hidnorm[b] = hn; }

    float mvW[4], mvU[4], t1[4], t2[4], g[4];

    // z gate
    matvec_scale4(wzv, hn, mvW, smem);
    matvec_scale4(uzv, xn, mvU, smem);
    mobius_add4(mvW, mvU, t1, smem);
    mobius_add4(t1, bzv, t2, smem);
    logmap0_4(t2, g, smem);
    float zv[4];
#pragma unroll
    for (int i = 0; i < 4; ++i) zv[i] = 1.f / (1.f + expf(-g[i]));
    store4(z_out + base + i0, zv);

    // r gate
    matvec_scale4(wrv, hn, mvW, smem);
    matvec_scale4(urv, xn, mvU, smem);
    mobius_add4(mvW, mvU, t1, smem);
    mobius_add4(t1, brv, t2, smem);
    logmap0_4(t2, g, smem);
    float rv[4];
#pragma unroll
    for (int i = 0; i < 4; ++i) rv[i] = 1.f / (1.f + expf(-g[i]));

    // r (.) h  -> bf16 for the dependent GEMM
    float rph[4];
    float rn = pointwise_mul4(hv, rv, rph, smem);
    if (threadIdx.x == 0) rphnorm[b] = fmaxf(rn, EPSF);
#pragma unroll
    for (int i = 0; i < 4; ++i) rph_bf[base + i0 + i] = f2bf_rne(rph[i]);
}

// ---------------------------------------------------------------- final combine

__global__ __launch_bounds__(256) void final_kernel(
        const float* __restrict__ hid, const float* __restrict__ whh,
        const float* __restrict__ uxh, const float* __restrict__ bh,
        const float* __restrict__ z, const float* __restrict__ xnorm,
        const float* __restrict__ rphnorm, float* __restrict__ out) {
    __shared__ float smem[8];
    const int    b    = blockIdx.x;
    const size_t base = (size_t)b * 1024;
    const int    i0   = threadIdx.x * 4;

    float hv[4], whv[4], uxv[4], bhv[4], zv[4];
    load4(hv,  hid + base + i0);
    load4(whv, whh + base + i0);
    load4(uxv, uxh + base + i0);
    load4(bhv, bh + i0);
    load4(zv,  z + base + i0);

    const float xn = xnorm[b];
    const float rn = rphnorm[b];

    float mvW[4], mvU[4], t1[4], ht[4];
    matvec_scale4(whv, rn, mvW, smem);
    matvec_scale4(uxv, xn, mvU, smem);
    mobius_add4(mvW, mvU, t1, smem);
    mobius_add4(t1, bhv, ht, smem);

    float nh[4];
#pragma unroll
    for (int i = 0; i < 4; ++i) nh[i] = -hv[i];
    float mho[4];
    mobius_add4(nh, ht, mho, smem);

    float p[4];
    pointwise_mul4(mho, zv, p, smem);

    float res[4];
    mobius_add4(hv, p, res, smem);
    store4(out + base + i0, res);
}

// ---------------------------------------------------------------- launch

extern "C" void kernel_launch(void* const* d_in, const int* in_sizes, int n_in,
                              void* d_out, int out_size, void* d_ws, size_t ws_size,
                              hipStream_t stream) {
    (void)in_sizes; (void)n_in; (void)out_size; (void)ws_size;
    constexpr int B = 8192, D = 1024;
    constexpr size_t MB = 1u << 20;

    const float* hyp_x  = (const float*)d_in[0];
    const float* hidden = (const float*)d_in[1];
    const float* w_z    = (const float*)d_in[2];
    const float* w_r    = (const float*)d_in[3];
    const float* w_h    = (const float*)d_in[4];
    const float* u_z    = (const float*)d_in[5];
    const float* u_r    = (const float*)d_in[6];
    const float* u_h    = (const float*)d_in[7];
    const float* b_z    = (const float*)d_in[8];
    const float* b_r    = (const float*)d_in[9];
    const float* b_h    = (const float*)d_in[10];
    float* out = (float*)d_out;
    char*  ws  = (char*)d_ws;

    // bf16 staging
    ushort_t* xbf   = (ushort_t*)(ws + 0);        // 16 MB
    ushort_t* hbf   = (ushort_t*)(ws + 16 * MB);  // 16 MB
    ushort_t* wzbf  = (ushort_t*)(ws + 32 * MB);  // 2 MB each
    ushort_t* wrbf  = (ushort_t*)(ws + 34 * MB);
    ushort_t* whbf  = (ushort_t*)(ws + 36 * MB);
    ushort_t* uzbf  = (ushort_t*)(ws + 38 * MB);
    ushort_t* urbf  = (ushort_t*)(ws + 40 * MB);
    ushort_t* uhbf  = (ushort_t*)(ws + 42 * MB);
    ushort_t* rphbf = (ushort_t*)(ws + 44 * MB);  // 16 MB
    // f32 GEMM outputs
    float* whzO = (float*)(ws + 64 * MB);
    float* uxzO = (float*)(ws + 96 * MB);
    float* whrO = (float*)(ws + 128 * MB);
    float* uxrO = (float*)(ws + 160 * MB);
    float* uxhO = (float*)(ws + 192 * MB);
    float* whhO = (float*)(ws + 224 * MB);
    float* zbuf = (float*)(ws + 256 * MB);
    // per-row norms
    float* xnorm   = (float*)(ws + 288 * MB);
    float* hidnorm = xnorm + B;
    float* rphnorm = xnorm + 2 * B;

    const int nBD = B * D;       // 8388608
    const int nDD = D * D;       // 1048576

    cvt_f32_bf16<<<nBD / 256, 256, 0, stream>>>(hyp_x, xbf, nBD);
    cvt_f32_bf16<<<nBD / 256, 256, 0, stream>>>(hidden, hbf, nBD);
    cvt_f32_bf16<<<nDD / 256, 256, 0, stream>>>(w_z, wzbf, nDD);
    cvt_f32_bf16<<<nDD / 256, 256, 0, stream>>>(w_r, wrbf, nDD);
    cvt_f32_bf16<<<nDD / 256, 256, 0, stream>>>(w_h, whbf, nDD);
    cvt_f32_bf16<<<nDD / 256, 256, 0, stream>>>(u_z, uzbf, nDD);
    cvt_f32_bf16<<<nDD / 256, 256, 0, stream>>>(u_r, urbf, nDD);
    cvt_f32_bf16<<<nDD / 256, 256, 0, stream>>>(u_h, uhbf, nDD);

    dim3 gg(D / 128, B / 256);   // (8, 32)
    gemm_bf16_nt<<<gg, 256, 0, stream>>>(hbf, wzbf, whzO);
    gemm_bf16_nt<<<gg, 256, 0, stream>>>(hbf, wrbf, whrO);
    gemm_bf16_nt<<<gg, 256, 0, stream>>>(xbf, uzbf, uxzO);
    gemm_bf16_nt<<<gg, 256, 0, stream>>>(xbf, urbf, uxrO);
    gemm_bf16_nt<<<gg, 256, 0, stream>>>(xbf, uhbf, uxhO);

    gates_kernel<<<B, 256, 0, stream>>>(hyp_x, hidden, whzO, uxzO, whrO, uxrO,
                                        b_z, b_r, zbuf, rphbf,
                                        xnorm, hidnorm, rphnorm);

    gemm_bf16_nt<<<gg, 256, 0, stream>>>(rphbf, whbf, whhO);

    final_kernel<<<B, 256, 0, stream>>>(hidden, whhO, uxhO, b_h, zbuf,
                                        xnorm, rphnorm, out);
}
